// NBodyGNN_80479097192830
// MI455X (gfx1250) — compile-verified, matrix operand order checked by reading, not run on hardware
//
#include <hip/hip_runtime.h>
#include <hip/hip_bf16.h>
#include <math.h>

// ---------------------------------------------------------------------------
// NBody GNN forward for MI455X (gfx1250, wave32, WMMA).
// Edge/Node MLPs run on v_wmma_f32_16x16x32_f16 (f16 in, f32 accumulate).
// ---------------------------------------------------------------------------

typedef __attribute__((ext_vector_type(16))) _Float16 v16h;
typedef __attribute__((ext_vector_type(8)))  float    v8f;

#define HD 128          // hidden size H
#define NLAYERS 3

union FragH { v16h v; uint4 q[2]; };

// K index inside a 16x32 f16 A (or prepacked B) fragment for (lane, half i)
__device__ __forceinline__ int frag_k(int lane, int i) {
    int ko = (lane < 16) ? 0 : 8;
    return (i < 8) ? (ko + i) : (16 + ko + (i - 8));
}

__device__ __forceinline__ float silu_f(float x) {
    return x / (1.0f + __expf(-x));
}

// ---------------------------------------------------------------------------
// Pack a (srcK x 128) fp32 weight matrix into f16 WMMA-B fragment order.
// Fragment (t, ks) covers output cols [16t,16t+16) and K rows [32ks,32ks+32).
// Each fragment = 512 halves; lane l owns 16 contiguous halves (32B).
// grid = 8*KS blocks of 32 threads; blockIdx = t*KS + ks.
// ---------------------------------------------------------------------------
__global__ void pack_b_kernel(const float* __restrict__ src,
                              _Float16* __restrict__ dst,
                              int srcK, int KS) {
    int b    = blockIdx.x;
    int ks   = b % KS;
    int lane = threadIdx.x;
    int n    = ((b / KS) * 16) + (lane & 15);
    _Float16* d = dst + (size_t)b * 512 + (size_t)lane * 16;
    for (int i = 0; i < 16; ++i) {
        int k = 32 * ks + frag_k(lane, i);
        float v = (k < srcK) ? src[(size_t)k * HD + n] : 0.0f;
        d[i] = (_Float16)v;
    }
}

// ---------------------------------------------------------------------------
// Encoder: h = (silu(LN(x@w1+b1)))@w2 + b2      one block (128 thr) per node
// ---------------------------------------------------------------------------
__global__ void encoder_kernel(const float* __restrict__ x,
                               const float* __restrict__ w1, const float* __restrict__ b1,
                               const float* __restrict__ g,  const float* __restrict__ be,
                               const float* __restrict__ w2, const float* __restrict__ b2,
                               float* __restrict__ hout) {
    int node = blockIdx.x;
    int tid  = threadIdx.x;
    __shared__ float xr[7];
    __shared__ float hid[HD];
    __shared__ float red[HD];

    if (tid < 7) xr[tid] = x[(size_t)node * 7 + tid];
    __syncthreads();

    float a = b1[tid];
    #pragma unroll
    for (int k = 0; k < 7; ++k) a += xr[k] * w1[k * HD + tid];

    red[tid] = a; __syncthreads();
    for (int s = 64; s > 0; s >>= 1) { if (tid < s) red[tid] += red[tid + s]; __syncthreads(); }
    float m = red[0] * (1.0f / HD); __syncthreads();
    float d = a - m;
    red[tid] = d * d; __syncthreads();
    for (int s = 64; s > 0; s >>= 1) { if (tid < s) red[tid] += red[tid + s]; __syncthreads(); }
    float inv = rsqrtf(red[0] * (1.0f / HD) + 1e-5f); __syncthreads();

    hid[tid] = silu_f(d * inv * g[tid] + be[tid]);
    __syncthreads();

    float o = b2[tid];
    for (int k = 0; k < HD; ++k) o += hid[k] * w2[k * HD + tid];
    hout[(size_t)node * HD + tid] = o;
}

// ---------------------------------------------------------------------------
// Edge MLP + scatter-add.  2 waves per block; each wave owns 16 edges.
//   A = [h[col] | h[row] | edge_attr(5)]  (K padded 261 -> 288)
//   msg = (silu(LN(A@w1+b1)))@w2 + b2 ; aggr[col] += msg  (fp32 atomics)
// ---------------------------------------------------------------------------
__launch_bounds__(64)
__global__ void edge_mlp_kernel(const int* __restrict__ ei,     // [2*E]
                                const float* __restrict__ pos,  // [N*3]
                                const float* __restrict__ h,    // [N*128]
                                float* __restrict__ aggr,       // [N*128]
                                const _Float16* __restrict__ pw1, // packed 8x9 frags
                                const _Float16* __restrict__ pw2, // packed 8x4 frags
                                const float* __restrict__ b1, const float* __restrict__ gg,
                                const float* __restrict__ be, const float* __restrict__ b2,
                                int E_) {
    __shared__ __align__(16) _Float16 Abuf[2][16][296];   // K stride 296 (288 used)
    __shared__ __align__(16) float    Hid [2][16][HD];
    __shared__ __align__(16) _Float16 A2  [2][16][HD];
    __shared__ int colIdx[2][16];

    const int lane = threadIdx.x & 31;
    const int w    = threadIdx.x >> 5;
    const int e0   = (blockIdx.x * 2 + w) * 16;

    // ---- gather h rows as f16, build physics features ----
    for (int r = 0; r < 16; ++r) {
        int e = e0 + r; if (e >= E_) e = E_ - 1;
        int ci = ei[(size_t)E_ + e];   // col / target i
        int rj = ei[e];                // row / source j
        float4 av = ((const float4*)(h + (size_t)ci * HD))[lane];
        float4 bv = ((const float4*)(h + (size_t)rj * HD))[lane];
        _Float16* ar = &Abuf[w][r][0];
        ar[lane * 4 + 0] = (_Float16)av.x;  ar[lane * 4 + 1] = (_Float16)av.y;
        ar[lane * 4 + 2] = (_Float16)av.z;  ar[lane * 4 + 3] = (_Float16)av.w;
        ar[HD + lane * 4 + 0] = (_Float16)bv.x;  ar[HD + lane * 4 + 1] = (_Float16)bv.y;
        ar[HD + lane * 4 + 2] = (_Float16)bv.z;  ar[HD + lane * 4 + 3] = (_Float16)bv.w;
        if (lane == 0) colIdx[w][r] = ci;
    }
    if (lane < 16) {
        int r = lane;
        int e = e0 + r; if (e >= E_) e = E_ - 1;
        int ci = ei[(size_t)E_ + e];
        int rj = ei[e];
        float dx = pos[ci * 3 + 0] - pos[rj * 3 + 0];
        float dy = pos[ci * 3 + 1] - pos[rj * 3 + 1];
        float dz = pos[ci * 3 + 2] - pos[rj * 3 + 2];
        float dist = sqrtf(dx * dx + dy * dy + dz * dz) + 1e-8f;
        float rinv = 1.0f / dist;
        _Float16* ar = &Abuf[w][r][0];
        ar[256] = (_Float16)dist;
        ar[257] = (_Float16)(dx * rinv);
        ar[258] = (_Float16)(dy * rinv);
        ar[259] = (_Float16)(dz * rinv);
        ar[260] = (_Float16)(1.0f / (dist * dist + 1e-6f));
        for (int k = 261; k < 296; ++k) ar[k] = (_Float16)0.0f;
    }
    __syncthreads();

    const int n0 = lane & 15;
    const int mo = (lane >> 4) << 3;
    const int ko = (lane < 16) ? 0 : 8;
    const _Float16* arow = &Abuf[w][n0][0];

    // ---- G1: (16x288) @ (288x128) ----
    v8f acc[8] = {};
    #pragma unroll
    for (int ks = 0; ks < 9; ++ks) {
        FragH af;
        af.q[0] = *(const uint4*)(arow + 32 * ks + ko);
        af.q[1] = *(const uint4*)(arow + 32 * ks + 16 + ko);
        #pragma unroll
        for (int t = 0; t < 8; ++t) {
            FragH bf;
            const uint4* bp = (const uint4*)(pw1 + ((size_t)(t * 9 + ks)) * 512 + lane * 16);
            bf.q[0] = bp[0]; bf.q[1] = bp[1];
            acc[t] = __builtin_amdgcn_wmma_f32_16x16x32_f16(false, af.v, false, bf.v,
                                                            (short)0, acc[t], false, false);
        }
    }
    #pragma unroll
    for (int t = 0; t < 8; ++t)
        #pragma unroll
        for (int r = 0; r < 8; ++r)
            Hid[w][mo + r][t * 16 + n0] = acc[t][r];
    __syncthreads();

    // ---- bias + LayerNorm + SiLU (one lane per row) ----
    if (lane < 16) {
        int row = lane;
        float m = 0.0f;
        for (int n = 0; n < HD; ++n) { float v = Hid[w][row][n] + b1[n]; Hid[w][row][n] = v; m += v; }
        m *= (1.0f / HD);
        float var = 0.0f;
        for (int n = 0; n < HD; ++n) { float d = Hid[w][row][n] - m; var += d * d; }
        float inv = rsqrtf(var * (1.0f / HD) + 1e-5f);
        for (int n = 0; n < HD; ++n) {
            float v = (Hid[w][row][n] - m) * inv * gg[n] + be[n];
            A2[w][row][n] = (_Float16)silu_f(v);
        }
    }
    __syncthreads();

    // ---- G2: (16x128) @ (128x128) ----
    v8f acc2[8] = {};
    const _Float16* arow2 = &A2[w][n0][0];
    #pragma unroll
    for (int ks = 0; ks < 4; ++ks) {
        FragH af;
        af.q[0] = *(const uint4*)(arow2 + 32 * ks + ko);
        af.q[1] = *(const uint4*)(arow2 + 32 * ks + 16 + ko);
        #pragma unroll
        for (int t = 0; t < 8; ++t) {
            FragH bf;
            const uint4* bp = (const uint4*)(pw2 + ((size_t)(t * 4 + ks)) * 512 + lane * 16);
            bf.q[0] = bp[0]; bf.q[1] = bp[1];
            acc2[t] = __builtin_amdgcn_wmma_f32_16x16x32_f16(false, af.v, false, bf.v,
                                                             (short)0, acc2[t], false, false);
        }
    }

    // ---- scatter-add messages into aggr (segment_sum by col) ----
    #pragma unroll
    for (int t = 0; t < 8; ++t) {
        int n = t * 16 + n0;
        float bb = b2[n];
        #pragma unroll
        for (int r = 0; r < 8; ++r) {
            int m = mo + r;
            if (e0 + m < E_)
                atomicAdd(&aggr[(size_t)colIdx[w][m] * HD + n], acc2[t][r] + bb);
        }
    }
}

// ---------------------------------------------------------------------------
// Node update:  h' = LN( h + MLP2([h | aggr]) )    2 waves/block, 16 nodes/wave
// ---------------------------------------------------------------------------
__launch_bounds__(64)
__global__ void node_mlp_kernel(const float* __restrict__ h,
                                const float* __restrict__ aggr,
                                float* __restrict__ hnext,
                                const _Float16* __restrict__ pw1, // 8x8 frags
                                const _Float16* __restrict__ pw2, // 8x4 frags
                                const float* __restrict__ b1, const float* __restrict__ gg,
                                const float* __restrict__ be, const float* __restrict__ b2,
                                const float* __restrict__ lng, const float* __restrict__ lnb,
                                int N_) {
    __shared__ __align__(16) _Float16 Abuf[2][16][264];  // K stride 264 (256 used)
    __shared__ __align__(16) float    Hid [2][16][HD];
    __shared__ __align__(16) _Float16 A2  [2][16][HD];

    const int lane = threadIdx.x & 31;
    const int w    = threadIdx.x >> 5;
    const int i0   = (blockIdx.x * 2 + w) * 16;

    for (int r = 0; r < 16; ++r) {
        int node = i0 + r; if (node >= N_) node = N_ - 1;
        float4 av = ((const float4*)(h    + (size_t)node * HD))[lane];
        float4 gv = ((const float4*)(aggr + (size_t)node * HD))[lane];
        _Float16* ar = &Abuf[w][r][0];
        ar[lane * 4 + 0] = (_Float16)av.x;  ar[lane * 4 + 1] = (_Float16)av.y;
        ar[lane * 4 + 2] = (_Float16)av.z;  ar[lane * 4 + 3] = (_Float16)av.w;
        ar[HD + lane * 4 + 0] = (_Float16)gv.x;  ar[HD + lane * 4 + 1] = (_Float16)gv.y;
        ar[HD + lane * 4 + 2] = (_Float16)gv.z;  ar[HD + lane * 4 + 3] = (_Float16)gv.w;
    }
    if (lane < 16)
        for (int k = 256; k < 264; ++k) Abuf[w][lane][k] = (_Float16)0.0f;
    __syncthreads();

    const int n0 = lane & 15;
    const int mo = (lane >> 4) << 3;
    const int ko = (lane < 16) ? 0 : 8;
    const _Float16* arow = &Abuf[w][n0][0];

    v8f acc[8] = {};
    #pragma unroll
    for (int ks = 0; ks < 8; ++ks) {
        FragH af;
        af.q[0] = *(const uint4*)(arow + 32 * ks + ko);
        af.q[1] = *(const uint4*)(arow + 32 * ks + 16 + ko);
        #pragma unroll
        for (int t = 0; t < 8; ++t) {
            FragH bf;
            const uint4* bp = (const uint4*)(pw1 + ((size_t)(t * 8 + ks)) * 512 + lane * 16);
            bf.q[0] = bp[0]; bf.q[1] = bp[1];
            acc[t] = __builtin_amdgcn_wmma_f32_16x16x32_f16(false, af.v, false, bf.v,
                                                            (short)0, acc[t], false, false);
        }
    }
    #pragma unroll
    for (int t = 0; t < 8; ++t)
        #pragma unroll
        for (int r = 0; r < 8; ++r)
            Hid[w][mo + r][t * 16 + n0] = acc[t][r];
    __syncthreads();

    if (lane < 16) {
        int row = lane;
        float m = 0.0f;
        for (int n = 0; n < HD; ++n) { float v = Hid[w][row][n] + b1[n]; Hid[w][row][n] = v; m += v; }
        m *= (1.0f / HD);
        float var = 0.0f;
        for (int n = 0; n < HD; ++n) { float d = Hid[w][row][n] - m; var += d * d; }
        float inv = rsqrtf(var * (1.0f / HD) + 1e-5f);
        for (int n = 0; n < HD; ++n) {
            float v = (Hid[w][row][n] - m) * inv * gg[n] + be[n];
            A2[w][row][n] = (_Float16)silu_f(v);
        }
    }
    __syncthreads();

    v8f acc2[8] = {};
    const _Float16* arow2 = &A2[w][n0][0];
    #pragma unroll
    for (int ks = 0; ks < 4; ++ks) {
        FragH af;
        af.q[0] = *(const uint4*)(arow2 + 32 * ks + ko);
        af.q[1] = *(const uint4*)(arow2 + 32 * ks + 16 + ko);
        #pragma unroll
        for (int t = 0; t < 8; ++t) {
            FragH bf;
            const uint4* bp = (const uint4*)(pw2 + ((size_t)(t * 4 + ks)) * 512 + lane * 16);
            bf.q[0] = bp[0]; bf.q[1] = bp[1];
            acc2[t] = __builtin_amdgcn_wmma_f32_16x16x32_f16(false, af.v, false, bf.v,
                                                             (short)0, acc2[t], false, false);
        }
    }
    __syncthreads();   // Hid reuse below
    #pragma unroll
    for (int t = 0; t < 8; ++t)
        #pragma unroll
        for (int r = 0; r < 8; ++r)
            Hid[w][mo + r][t * 16 + n0] = acc2[t][r];
    __syncthreads();

    // residual + final LayerNorm, write h'
    if (lane < 16) {
        int row = lane;
        int node = i0 + row;
        if (node < N_) {
            float m = 0.0f;
            for (int n = 0; n < HD; ++n) {
                float v = Hid[w][row][n] + b2[n] + h[(size_t)node * HD + n];
                Hid[w][row][n] = v; m += v;
            }
            m *= (1.0f / HD);
            float var = 0.0f;
            for (int n = 0; n < HD; ++n) { float d = Hid[w][row][n] - m; var += d * d; }
            float inv = rsqrtf(var * (1.0f / HD) + 1e-5f);
            for (int n = 0; n < HD; ++n)
                hnext[(size_t)node * HD + n] = (Hid[w][row][n] - m) * inv * lng[n] + lnb[n];
        }
    }
}

// ---------------------------------------------------------------------------
// Decoder: out = x[:, :6] + (silu(silu(h@w1+b1)@w2+b2))@w3 + b3
// ---------------------------------------------------------------------------
__global__ void decoder_kernel(const float* __restrict__ x, const float* __restrict__ h,
                               const float* __restrict__ w1, const float* __restrict__ b1,
                               const float* __restrict__ w2, const float* __restrict__ b2,
                               const float* __restrict__ w3, const float* __restrict__ b3,
                               float* __restrict__ out) {
    int node = blockIdx.x;
    int tid  = threadIdx.x;
    __shared__ float hrow[HD];
    __shared__ float d1[HD];
    __shared__ float d2[64];

    hrow[tid] = h[(size_t)node * HD + tid];
    __syncthreads();
    float a = b1[tid];
    for (int k = 0; k < HD; ++k) a += hrow[k] * w1[k * HD + tid];
    d1[tid] = silu_f(a);
    __syncthreads();
    if (tid < 64) {
        float a2 = b2[tid];
        for (int k = 0; k < HD; ++k) a2 += d1[k] * w2[k * 64 + tid];
        d2[tid] = silu_f(a2);
    }
    __syncthreads();
    if (tid < 6) {
        float o = b3[tid];
        for (int k = 0; k < 64; ++k) o += d2[k] * w3[k * 6 + tid];
        out[(size_t)node * 6 + tid] = x[(size_t)node * 7 + tid] + o;
    }
}

__global__ void zero_kernel(float* __restrict__ p, size_t n) {
    size_t i = (size_t)blockIdx.x * blockDim.x + threadIdx.x;
    size_t stride = (size_t)gridDim.x * blockDim.x;
    for (; i < n; i += stride) p[i] = 0.0f;
}

// ---------------------------------------------------------------------------
extern "C" void kernel_launch(void* const* d_in, const int* in_sizes, int n_in,
                              void* d_out, int out_size, void* d_ws, size_t ws_size,
                              hipStream_t stream) {
    const float* x      = (const float*)d_in[0];
    const float* pos    = (const float*)d_in[1];
    const int*   ei     = (const int*)d_in[2];      // [2*E] (int32)
    const float* enc_w1 = (const float*)d_in[3];
    const float* enc_b1 = (const float*)d_in[4];
    const float* enc_g  = (const float*)d_in[5];
    const float* enc_be = (const float*)d_in[6];
    const float* enc_w2 = (const float*)d_in[7];
    const float* enc_b2 = (const float*)d_in[8];
    const float* e_w1   = (const float*)d_in[9];    // (3,261,128)
    const float* e_b1   = (const float*)d_in[10];
    const float* e_g    = (const float*)d_in[11];
    const float* e_be   = (const float*)d_in[12];
    const float* e_w2   = (const float*)d_in[13];   // (3,128,128)
    const float* e_b2   = (const float*)d_in[14];
    const float* n_w1   = (const float*)d_in[15];   // (3,256,128)
    const float* n_b1   = (const float*)d_in[16];
    const float* n_g    = (const float*)d_in[17];
    const float* n_be   = (const float*)d_in[18];
    const float* n_w2   = (const float*)d_in[19];   // (3,128,128)
    const float* n_b2   = (const float*)d_in[20];
    const float* ln_g   = (const float*)d_in[21];
    const float* ln_b   = (const float*)d_in[22];
    const float* dec_w1 = (const float*)d_in[23];
    const float* dec_b1 = (const float*)d_in[24];
    const float* dec_w2 = (const float*)d_in[25];
    const float* dec_b2 = (const float*)d_in[26];
    const float* dec_w3 = (const float*)d_in[27];
    const float* dec_b3 = (const float*)d_in[28];

    const int N_ = in_sizes[0] / 7;
    const int E_ = in_sizes[2] / 2;

    // workspace carve-out (256B aligned)
    char* ws = (char*)d_ws;
    size_t off = 0;
    auto carve = [&](size_t bytes) -> void* {
        void* p = ws + off;
        off = (off + bytes + 255) & ~(size_t)255;
        return p;
    };
    float*     hA   = (float*)carve((size_t)N_ * HD * sizeof(float));
    float*     hB   = (float*)carve((size_t)N_ * HD * sizeof(float));
    float*     aggr = (float*)carve((size_t)N_ * HD * sizeof(float));
    _Float16*  pe1  = (_Float16*)carve((size_t)NLAYERS * 72 * 512 * 2); // 8 Ntiles x 9 Ksteps
    _Float16*  pe2  = (_Float16*)carve((size_t)NLAYERS * 32 * 512 * 2); // 8 x 4
    _Float16*  pn1  = (_Float16*)carve((size_t)NLAYERS * 64 * 512 * 2); // 8 x 8
    _Float16*  pn2  = (_Float16*)carve((size_t)NLAYERS * 32 * 512 * 2); // 8 x 4
    (void)ws_size; (void)n_in; (void)out_size;

    // --- weight prep (f16, B-fragment order) ---
    for (int l = 0; l < NLAYERS; ++l) {
        pack_b_kernel<<<8 * 9, 32, 0, stream>>>(e_w1 + (size_t)l * 261 * HD, pe1 + (size_t)l * 72 * 512, 261, 9);
        pack_b_kernel<<<8 * 4, 32, 0, stream>>>(e_w2 + (size_t)l * HD * HD,  pe2 + (size_t)l * 32 * 512, 128, 4);
        pack_b_kernel<<<8 * 8, 32, 0, stream>>>(n_w1 + (size_t)l * 256 * HD, pn1 + (size_t)l * 64 * 512, 256, 8);
        pack_b_kernel<<<8 * 4, 32, 0, stream>>>(n_w2 + (size_t)l * HD * HD,  pn2 + (size_t)l * 32 * 512, 128, 4);
    }

    // --- encoder ---
    encoder_kernel<<<N_, 128, 0, stream>>>(x, enc_w1, enc_b1, enc_g, enc_be, enc_w2, enc_b2, hA);

    // --- message-passing layers ---
    float* hc = hA;
    float* hn = hB;
    const int edge_blocks = (E_ + 31) / 32;   // 2 tiles of 16 edges per block
    const int node_blocks = (N_ + 31) / 32;
    for (int l = 0; l < NLAYERS; ++l) {
        zero_kernel<<<2048, 256, 0, stream>>>(aggr, (size_t)N_ * HD);
        edge_mlp_kernel<<<edge_blocks, 64, 0, stream>>>(
            ei, pos, hc, aggr,
            pe1 + (size_t)l * 72 * 512, pe2 + (size_t)l * 32 * 512,
            e_b1 + l * HD, e_g + l * HD, e_be + l * HD, e_b2 + l * HD, E_);
        node_mlp_kernel<<<node_blocks, 64, 0, stream>>>(
            hc, aggr, hn,
            pn1 + (size_t)l * 64 * 512, pn2 + (size_t)l * 32 * 512,
            n_b1 + l * HD, n_g + l * HD, n_be + l * HD, n_b2 + l * HD,
            ln_g + l * HD, ln_b + l * HD, N_);
        float* tmp = hc; hc = hn; hn = tmp;
    }

    // --- decoder ---
    decoder_kernel<<<N_, 128, 0, stream>>>(x, hc, dec_w1, dec_b1, dec_w2, dec_b2,
                                           dec_w3, dec_b3, (float*)d_out);
}
